// GraphSage_69140383531488
// MI455X (gfx1250) — compile-verified, matrix operand order checked by reading, not run on hardware
//
#include <hip/hip_runtime.h>

// ---------------------------------------------------------------------------
// CDNA5 (gfx1250) GraphSAGE forward: fp32 means + bf16 WMMA GEMMs (f32 acc).
// D=512, B=8192, K=8, C=64.
// ---------------------------------------------------------------------------

typedef __attribute__((ext_vector_type(16))) __bf16 v16bf;
typedef __attribute__((ext_vector_type(8)))  float  v8f;

union BV {
    uint4  u[2];
    v16bf  v;
};

static __device__ inline unsigned int bf16_bits(float f) {
    union { __bf16 b; unsigned short s; } cv;
    cv.b = (__bf16)f;                       // RNE convert
    return (unsigned int)cv.s;
}
static __device__ inline float bf16u_to_f(unsigned int bits16) {
    union { unsigned int u; float f; } cv;
    cv.u = bits16 << 16;
    return cv.f;
}

// ---------------------------------------------------------------------------
// Pack an fp32 weight [Kdim, N] into the CDNA5 bf16 B-operand tile layout.
// Tile (nt, kt) = 32 lanes x 8 dwords = 1KB, stored contiguously:
//   word(tid): t = tid>>8 (tile), l = (tid>>3)&31 (lane), v = tid&7 (dword)
//   n  = nt*16 + (l&15),  h = l>>4
//   k0 = kt*32 + (v>>2)*16 + h*8 + (v&3)*2 ; word = {bf16(w[k0+1,n]), bf16(w[k0,n])}
// ---------------------------------------------------------------------------
__global__ void pack_weight(const float* __restrict__ w, unsigned int* __restrict__ dst,
                            int Kdim, int N) {
    int tid = blockIdx.x * blockDim.x + threadIdx.x;
    int total = (Kdim * N) >> 1;
    if (tid >= total) return;
    int t   = tid >> 8;
    int rem = tid & 255;
    int l   = rem >> 3;
    int v   = rem & 7;
    int ktiles = Kdim >> 5;
    int nt = t / ktiles;
    int kt = t - nt * ktiles;
    int n  = (nt << 4) + (l & 15);
    int h  = l >> 4;
    int k0 = (kt << 5) + ((v >> 2) << 4) + (h << 3) + ((v & 3) << 1);
    unsigned int lo = bf16_bits(w[k0 * N + n]);
    unsigned int hi = bf16_bits(w[(k0 + 1) * N + n]);
    dst[tid] = (hi << 16) | lo;
}

// ---------------------------------------------------------------------------
// agg[r, :] = mean_k src[r*8+k, :]   (fp32 in, bf16 out, 4 elements/thread)
// ---------------------------------------------------------------------------
__global__ void mean8_f32_to_bf16(const float* __restrict__ src, uint2* __restrict__ dst,
                                  int R, int D) {
    long long tid = (long long)blockIdx.x * blockDim.x + threadIdx.x;
    long long total4 = ((long long)R * D) >> 2;
    if (tid >= total4) return;
    long long base = tid << 2;
    int r = (int)(base / D);
    int c = (int)(base - (long long)r * D);
    float ax = 0.f, ay = 0.f, az = 0.f, aw = 0.f;
#pragma unroll
    for (int k = 0; k < 8; ++k) {
        float4 p = *(const float4*)(src + ((long long)(r * 8 + k) * D + c));
        ax += p.x; ay += p.y; az += p.z; aw += p.w;
    }
    uint2 o;
    o.x = (bf16_bits(ay * 0.125f) << 16) | bf16_bits(ax * 0.125f);
    o.y = (bf16_bits(aw * 0.125f) << 16) | bf16_bits(az * 0.125f);
    dst[tid] = o;
}

// ---------------------------------------------------------------------------
// agg[r, :] = mean_k src[r*8+k, :]   (bf16 in, bf16 out, 4 elements/thread)
// ---------------------------------------------------------------------------
__global__ void mean8_bf16(const unsigned int* __restrict__ src, uint2* __restrict__ dst,
                           int R, int D) {
    long long tid = (long long)blockIdx.x * blockDim.x + threadIdx.x;
    long long total4 = ((long long)R * D) >> 2;
    if (tid >= total4) return;
    long long base = tid << 2;
    int r = (int)(base / D);
    int c = (int)(base - (long long)r * D);
    float a0 = 0.f, a1 = 0.f, a2 = 0.f, a3 = 0.f;
#pragma unroll
    for (int k = 0; k < 8; ++k) {
        uint2 p = *(const uint2*)(src + (((long long)(r * 8 + k) * D + c) >> 1));
        a0 += bf16u_to_f(p.x & 0xffffu); a1 += bf16u_to_f(p.x >> 16);
        a2 += bf16u_to_f(p.y & 0xffffu); a3 += bf16u_to_f(p.y >> 16);
    }
    uint2 o;
    o.x = (bf16_bits(a1 * 0.125f) << 16) | bf16_bits(a0 * 0.125f);
    o.y = (bf16_bits(a3 * 0.125f) << 16) | bf16_bits(a2 * 0.125f);
    dst[tid] = o;
}

// ---------------------------------------------------------------------------
// WMMA GEMM: out[R,N] = act( xcur @ Wb + (HAS_AGG ? xagg @ Wa : 0) + bias )
//   xcur: fp32 [R,Dk] if CUR_F32 else bf16; xagg: bf16 [R,Dk]
//   Wb/Wa: packed bf16 B-operand tiles (see pack_weight)
// Block = 256 threads (8 waves). Block tile = 64 rows x 64 cols.
// Wave tile = 16 rows x 32 cols (two f32 16x16 accumulators).
// ---------------------------------------------------------------------------
template<bool CUR_F32, bool HAS_AGG, bool RELU, bool OUT_BF16>
__global__ __launch_bounds__(256)
void sage_gemm(const void* __restrict__ xcur, const void* __restrict__ xagg,
               const unsigned int* __restrict__ wb, const unsigned int* __restrict__ wa,
               const float* __restrict__ bias, void* __restrict__ out,
               int R, int Dk, int N) {
    const int lane = threadIdx.x & 31;
    const int wave = threadIdx.x >> 5;
    const int m16  = lane & 15;
    const int h    = lane >> 4;
    const int rowbase = blockIdx.x * 64 + (wave >> 1) * 16;
    const int colbase = blockIdx.y * 64 + (wave & 1) * 32;
    const int ktiles  = Dk >> 5;
    const long long arow = (long long)(rowbase + m16) * Dk;
    const int nt0 = colbase >> 4;

    v8f acc0 = {};
    v8f acc1 = {};

    for (int kt = 0; kt < ktiles; ++kt) {
        const int koff = (kt << 5) + (h << 3);

        // ---- A operand (current node features), 16x32 bf16 tile ----
        BV a;
        if (CUR_F32) {
            const float* p = (const float*)xcur + arow + koff;
            float tmp[16];
            *(float4*)(&tmp[0])  = ((const float4*)p)[0];
            *(float4*)(&tmp[4])  = ((const float4*)p)[1];
            *(float4*)(&tmp[8])  = ((const float4*)(p + 16))[0];
            *(float4*)(&tmp[12]) = ((const float4*)(p + 16))[1];
#pragma unroll
            for (int i = 0; i < 16; ++i) a.v[i] = (__bf16)tmp[i];
        } else {
            const unsigned short* p = (const unsigned short*)xcur + arow + koff;
            a.u[0] = *(const uint4*)(p);
            a.u[1] = *(const uint4*)(p + 16);
        }

        // ---- B operands (pre-packed weight tiles), 32B/lane each ----
        BV b0, b1;
        const unsigned int* wb0 = wb + (((nt0 + 0) * ktiles + kt) << 8) + (lane << 3);
        const unsigned int* wb1 = wb + (((nt0 + 1) * ktiles + kt) << 8) + (lane << 3);
        b0.u[0] = *(const uint4*)(wb0); b0.u[1] = *(const uint4*)(wb0 + 4);
        b1.u[0] = *(const uint4*)(wb1); b1.u[1] = *(const uint4*)(wb1 + 4);

        acc0 = __builtin_amdgcn_wmma_f32_16x16x32_bf16(false, a.v, false, b0.v,
                                                       (short)0, acc0, false, false);
        acc1 = __builtin_amdgcn_wmma_f32_16x16x32_bf16(false, a.v, false, b1.v,
                                                       (short)0, acc1, false, false);

        if (HAS_AGG) {
            BV ag, c0, c1;
            const unsigned short* p = (const unsigned short*)xagg + arow + koff;
            ag.u[0] = *(const uint4*)(p);
            ag.u[1] = *(const uint4*)(p + 16);
            const unsigned int* wa0 = wa + (((nt0 + 0) * ktiles + kt) << 8) + (lane << 3);
            const unsigned int* wa1 = wa + (((nt0 + 1) * ktiles + kt) << 8) + (lane << 3);
            c0.u[0] = *(const uint4*)(wa0); c0.u[1] = *(const uint4*)(wa0 + 4);
            c1.u[0] = *(const uint4*)(wa1); c1.u[1] = *(const uint4*)(wa1 + 4);
            acc0 = __builtin_amdgcn_wmma_f32_16x16x32_bf16(false, ag.v, false, c0.v,
                                                           (short)0, acc0, false, false);
            acc1 = __builtin_amdgcn_wmma_f32_16x16x32_bf16(false, ag.v, false, c1.v,
                                                           (short)0, acc1, false, false);
        }
    }

    // ---- epilogue: C/D layout is row M = r + h*8, col N = lane&15 ----
    float bia0 = 0.f, bia1 = 0.f;
    if (bias) { bia0 = bias[colbase + m16]; bia1 = bias[colbase + 16 + m16]; }
    const int orow0 = rowbase + h * 8;
#pragma unroll
    for (int r = 0; r < 8; ++r) {
        const long long orow = (long long)(orow0 + r) * N;
        float x0 = acc0[r] + bia0;
        float x1 = acc1[r] + bia1;
        if (RELU) { x0 = fmaxf(x0, 0.f); x1 = fmaxf(x1, 0.f); }
        if (OUT_BF16) {
            unsigned short* o = (unsigned short*)out;
            o[orow + colbase + m16]      = (unsigned short)bf16_bits(x0);
            o[orow + colbase + 16 + m16] = (unsigned short)bf16_bits(x1);
        } else {
            float* o = (float*)out;
            o[orow + colbase + m16]      = x0;
            o[orow + colbase + 16 + m16] = x1;
        }
    }
}

// ---------------------------------------------------------------------------
// Host-side orchestration
// ---------------------------------------------------------------------------
extern "C" void kernel_launch(void* const* d_in, const int* in_sizes, int n_in,
                              void* d_out, int out_size, void* d_ws, size_t ws_size,
                              hipStream_t stream) {
    (void)in_sizes; (void)n_in; (void)out_size; (void)ws_size;

    constexpr int D  = 512;
    constexpr int B  = 8192;
    constexpr int K  = 8;
    constexpr int C  = 64;
    constexpr int H1 = D / 2;   // 256
    constexpr int H2 = D / 4;   // 128

    const float* nf0   = (const float*)d_in[0];   // [B, D]
    const float* nf1   = (const float*)d_in[1];   // [B*K, D]
    const float* nf2   = (const float*)d_in[2];   // [B*K*K, D]
    const float* w1_0  = (const float*)d_in[3];   // [D, H1]
    const float* w2_0  = (const float*)d_in[4];   // [D, H1]
    const float* w1_1  = (const float*)d_in[5];   // [H1, H2]
    const float* w2_1  = (const float*)d_in[6];   // [H1, H2]
    const float* cls_w = (const float*)d_in[7];   // [H2, C]
    const float* cls_b = (const float*)d_in[8];   // [C]
    float* out = (float*)d_out;                   // [B, C] fp32

    // -------- workspace carve-out (bf16 buffers + packed weights) --------
    char* ws = (char*)d_ws;
    size_t off = 0;
    auto carve = [&](size_t bytes) -> void* {
        void* p = ws + off;
        off += bytes;
        off = (off + 255) & ~(size_t)255;
        return p;
    };
    void* agg2  = carve((size_t)(B * K) * D  * 2);   // bf16 [B*K, D]
    void* agg1  = carve((size_t)B * D        * 2);   // bf16 [B,   D]
    void* a1    = carve((size_t)(B * K) * H1 * 2);   // bf16 [B*K, H1]
    void* a0    = carve((size_t)B * H1       * 2);   // bf16 [B,   H1]
    void* agga1 = carve((size_t)B * H1       * 2);   // bf16 [B,   H1]
    void* b0    = carve((size_t)B * H2       * 2);   // bf16 [B,   H2]
    unsigned int* w2_0p = (unsigned int*)carve((size_t)D  * H1 * 2);
    unsigned int* w1_0p = (unsigned int*)carve((size_t)D  * H1 * 2);
    unsigned int* w2_1p = (unsigned int*)carve((size_t)H1 * H2 * 2);
    unsigned int* w1_1p = (unsigned int*)carve((size_t)H1 * H2 * 2);
    unsigned int* clswp = (unsigned int*)carve((size_t)H2 * C  * 2);

    const int TB = 256;

    // -------- 1) pack weights into WMMA B-operand layout --------
    pack_weight<<<(D * H1 / 2 + TB - 1) / TB, TB, 0, stream>>>(w2_0,  w2_0p, D,  H1);
    pack_weight<<<(D * H1 / 2 + TB - 1) / TB, TB, 0, stream>>>(w1_0,  w1_0p, D,  H1);
    pack_weight<<<(H1 * H2 / 2 + TB - 1) / TB, TB, 0, stream>>>(w2_1,  w2_1p, H1, H2);
    pack_weight<<<(H1 * H2 / 2 + TB - 1) / TB, TB, 0, stream>>>(w1_1,  w1_1p, H1, H2);
    pack_weight<<<(H2 * C  / 2 + TB - 1) / TB, TB, 0, stream>>>(cls_w, clswp, H2, C);

    // -------- 2) neighbor means (fp32 accumulate, bf16 out) --------
    {
        long long t4 = (long long)(B * K) * D / 4;      // agg2 from nf2
        mean8_f32_to_bf16<<<(unsigned)((t4 + TB - 1) / TB), TB, 0, stream>>>(
            nf2, (uint2*)agg2, B * K, D);
    }
    {
        long long t4 = (long long)B * D / 4;            // agg1 from nf1
        mean8_f32_to_bf16<<<(unsigned)((t4 + TB - 1) / TB), TB, 0, stream>>>(
            nf1, (uint2*)agg1, B, D);
    }

    // -------- 3) layer-0 GEMMs: a1, a0 = relu(x@w2_0 + agg@w1_0) --------
    {
        dim3 grid(B * K / 64, H1 / 64);                 // a1: [B*K, 256]
        sage_gemm<true, true, true, true><<<grid, TB, 0, stream>>>(
            nf1, agg2, w2_0p, w1_0p, nullptr, a1, B * K, D, H1);
    }
    {
        dim3 grid(B / 64, H1 / 64);                     // a0: [B, 256]
        sage_gemm<true, true, true, true><<<grid, TB, 0, stream>>>(
            nf0, agg1, w2_0p, w1_0p, nullptr, a0, B, D, H1);
    }

    // -------- 4) mean of a1 over K neighbors --------
    {
        long long t4 = (long long)B * H1 / 4;
        mean8_bf16<<<(unsigned)((t4 + TB - 1) / TB), TB, 0, stream>>>(
            (const unsigned int*)a1, (uint2*)agga1, B, H1);
    }

    // -------- 5) layer-1 GEMM: b0 = relu(a0@w2_1 + agga1@w1_1) --------
    {
        dim3 grid(B / 64, H2 / 64);                     // b0: [B, 128]
        sage_gemm<false, true, true, true><<<grid, TB, 0, stream>>>(
            a0, agga1, w2_1p, w1_1p, nullptr, b0, B, H1, H2);
    }

    // -------- 6) classification head: out = b0@cls_w + cls_b (fp32) --------
    {
        dim3 grid(B / 64, C / 64);                      // out: [B, 64]
        sage_gemm<false, false, false, false><<<grid, TB, 0, stream>>>(
            b0, nullptr, clswp, nullptr, cls_b, out, B, H2, C);
    }
}
